// MultiHeadBAM_38783554683554
// MI455X (gfx1250) — matrix operand order repaired and checked by reading, not compiled, not run on hardware
//
#include <hip/hip_runtime.h>
#include <hip/hip_bf16.h>

typedef __attribute__((ext_vector_type(16))) __bf16 v16bf;
typedef __attribute__((ext_vector_type(8)))  float  v8f;

#define CDIM  2048
#define PDIM  256
#define BATCH 8
#define NTOT  (BATCH * PDIM)   // 2048 columns when batch is folded into N

// x / residual / outputs live in [B, C, P] layout; viewed as a C x (B*P) matrix:
__device__ __forceinline__ size_t bcp_idx(int c, int n) {
  return (size_t)(n >> 8) * (size_t)(CDIM * PDIM) + (size_t)c * PDIM + (n & (PDIM - 1));
}

// ---------------------------------------------------------------------------
// Y = gamma * (W @ X + bias) + resid      (gamma/resid optional)
// W: [2048 x 2048] row-major.  X, resid, Y: BCP layout (C x 2048 matrix view).
// fp32 emulated via bf16 hi/lo split: A*B ~= Ah*Bh + Ah*Bl + Al*Bh (f32 accum).
// ---------------------------------------------------------------------------
__global__ __launch_bounds__(256, 1)
void proj_gemm_bf16x3(const float* __restrict__ W,
                      const float* __restrict__ X,
                      const float* __restrict__ bias,
                      const float* __restrict__ gamma,   // may be null -> 1.0
                      const float* __restrict__ resid,   // may be null
                      float* __restrict__ Y)
{
  constexpr int BM = 128, BN = 128, BK = 32;
  // LDS tiles stored directly in WMMA fragment order: [tile][lane] -> 16 bf16
  __shared__ v16bf Ahi[BM / 16][32];
  __shared__ v16bf Alo[BM / 16][32];
  __shared__ v16bf Bhi[BN / 16][32];
  __shared__ v16bf Blo[BN / 16][32];

  const int tid  = threadIdx.x;
  const int lane = tid & 31;
  const int wid  = tid >> 5;          // 8 waves
  const int m0   = blockIdx.y * BM;
  const int n0   = blockIdx.x * BN;
  const int wmt  = (wid >> 1) * 2;    // wave's first mtile (0,2,4,6)
  const int wnt  = (wid & 1) * 4;     // wave's first ntile (0,4)

  v8f acc[2][4];
#pragma unroll
  for (int i = 0; i < 2; ++i)
#pragma unroll
    for (int j = 0; j < 4; ++j)
      acc[i][j] = (v8f)0.0f;

  // staging role: thread -> (tile st, lane) slot, for both A and B (8*32 = 256)
  const int st  = tid >> 5;
  const int rr  = lane & 15;            // row (A) / col (B) within 16-tile
  const int kbA = (lane >> 4) * 16;     // A fragment: lanes16-31 hold K base 16
  const int kbB = (lane >> 4) * 16;     // B fragment: lanes16-31 hold K base 16

  const float* wrow = W + (size_t)(m0 + st * 16 + rr) * CDIM;
  const int    colB = n0 + st * 16 + rr;

  for (int kc = 0; kc < CDIM; kc += BK) {
    // ---- stage A tile (BM x 32) in fragment order --------------------------
    {
      const float* p = wrow + kc + kbA;
      if (kc + BK < CDIM)
        __builtin_prefetch(p + BK, 0, 1);     // global_prefetch_b8 next chunk
      // A 16-bit 16x32 layout (contiguous): K = 16*(lane>>4) + e
      float4 f0 = *(const float4*)(p + 0);    // K = kbA+0..3
      float4 f1 = *(const float4*)(p + 4);    // K = kbA+4..7
      float4 f2 = *(const float4*)(p + 8);    // K = kbA+8..11
      float4 f3 = *(const float4*)(p + 12);   // K = kbA+12..15
      float fv[16] = {f0.x, f0.y, f0.z, f0.w, f1.x, f1.y, f1.z, f1.w,
                      f2.x, f2.y, f2.z, f2.w, f3.x, f3.y, f3.z, f3.w};
      v16bf hi, lo;
#pragma unroll
      for (int e = 0; e < 16; ++e) {
        __bf16 h = (__bf16)fv[e];
        hi[e] = h;
        lo[e] = (__bf16)(fv[e] - (float)h);
      }
      Ahi[st][lane] = hi;
      Alo[st][lane] = lo;
    }
    // ---- stage B tile (32 x BN) in fragment order --------------------------
    {
      v16bf hi, lo;
#pragma unroll
      for (int e = 0; e < 16; ++e) {
        float f = X[bcp_idx(kc + kbB + e, colB)];   // coalesced across lanes
        __bf16 h = (__bf16)f;
        hi[e] = h;
        lo[e] = (__bf16)(f - (float)h);
      }
      Bhi[st][lane] = hi;
      Blo[st][lane] = lo;
    }
    __syncthreads();

    // ---- WMMA compute: 2x4 tiles * 3 wmma each ----------------------------
    v16bf ah[2], al[2], bh[4], bl[4];
#pragma unroll
    for (int i = 0; i < 2; ++i) { ah[i] = Ahi[wmt + i][lane]; al[i] = Alo[wmt + i][lane]; }
#pragma unroll
    for (int j = 0; j < 4; ++j) { bh[j] = Bhi[wnt + j][lane]; bl[j] = Blo[wnt + j][lane]; }
#pragma unroll
    for (int i = 0; i < 2; ++i)
#pragma unroll
      for (int j = 0; j < 4; ++j) {
        acc[i][j] = __builtin_amdgcn_wmma_f32_16x16x32_bf16(
            false, ah[i], false, bh[j], (short)0, acc[i][j], false, false);
        acc[i][j] = __builtin_amdgcn_wmma_f32_16x16x32_bf16(
            false, ah[i], false, bl[j], (short)0, acc[i][j], false, false);
        acc[i][j] = __builtin_amdgcn_wmma_f32_16x16x32_bf16(
            false, al[i], false, bh[j], (short)0, acc[i][j], false, false);
      }
    __syncthreads();
  }

  // ---- epilogue: bias, optional gamma-scale + residual ---------------------
  const float g = gamma ? gamma[0] : 1.0f;
#pragma unroll
  for (int i = 0; i < 2; ++i)
#pragma unroll
    for (int j = 0; j < 4; ++j)
#pragma unroll
      for (int r = 0; r < 8; ++r) {
        // C/D layout: VGPR r -> M=r (lanes 0-15) / M=r+8 (lanes 16-31); N=lane&15
        int o = m0 + (wmt + i) * 16 + r + ((lane >> 4) << 3);
        int n = n0 + (wnt + j) * 16 + (lane & 15);
        float v = (acc[i][j][r] + bias[o]) * g;
        size_t idx = bcp_idx(o, n);
        if (resid) v += resid[idx];
        Y[idx] = v;
      }
}

// ---------------------------------------------------------------------------
// Fused energy -> softmax -> apply.  One WG per (batch, 4 consecutive d).
//   e_n[m]    = (1/16) * sum_h Q[b][h*256+n][d] * K[b][h*256+n][m]
//   a_n       = softmax_m(e_n)
//   out[h][m] = sum_n V[b][h*256+d][n] * a_n[m]
//   A[b][h*256+d][m] = out[h][m]
// Never materializes the 536 MB attn tensor.
// ---------------------------------------------------------------------------
__global__ __launch_bounds__(256, 1)
void attention_fused(const float* __restrict__ Q,
                     const float* __restrict__ K,
                     const float* __restrict__ V,
                     float* __restrict__ A)
{
  __shared__ float qs[4][8][256];        // q coefficient per (dt, h, n)
  __shared__ float vs[4][8][256];        // v coefficient per (dt, h, n)
  __shared__ float oa[2][4][8][256];     // per-n-half partial outputs

  const int b   = blockIdx.y;
  const int d0  = blockIdx.x * 4;
  const int tid = threadIdx.x;
  const size_t base = (size_t)b * (CDIM * PDIM);

  for (int i = tid; i < 2048; i += 256) {
    int h = i >> 8, n = i & 255;
    const float4 q4 = *(const float4*)(Q + base + (size_t)((h << 8) + n) * PDIM + d0);
    qs[0][h][n] = q4.x; qs[1][h][n] = q4.y; qs[2][h][n] = q4.z; qs[3][h][n] = q4.w;
  }
  for (int i = tid; i < 8192; i += 256) {
    int dt = i >> 11, h = (i >> 8) & 7, n = i & 255;
    vs[dt][h][n] = V[base + (size_t)((h << 8) + d0 + dt) * PDIM + n];
  }
  __syncthreads();

  const int lane = tid & 31;
  const int wid  = tid >> 5;
  const int dt   = wid >> 1;       // which of the 4 d-columns this wave owns
  const int nh   = wid & 1;        // which n-half

  float oacc[8][8];                // [h][j], lane owns columns m = lane*8 + j
#pragma unroll
  for (int h = 0; h < 8; ++h)
#pragma unroll
    for (int j = 0; j < 8; ++j) oacc[h][j] = 0.0f;

  for (int n = nh; n < 256; n += 2) {
    float e[8] = {0, 0, 0, 0, 0, 0, 0, 0};
#pragma unroll
    for (int h = 0; h < 8; ++h) {
      const float  qc = qs[dt][h][n];
      const float4* kr = (const float4*)(K + base + (size_t)((h << 8) + n) * PDIM) + lane * 2;
      float4 k0 = kr[0];
      float4 k1 = kr[1];
      e[0] += qc * k0.x; e[1] += qc * k0.y; e[2] += qc * k0.z; e[3] += qc * k0.w;
      e[4] += qc * k1.x; e[5] += qc * k1.y; e[6] += qc * k1.z; e[7] += qc * k1.w;
    }
    float mx = -3.402823466e38f;
#pragma unroll
    for (int j = 0; j < 8; ++j) { e[j] *= 0.0625f; mx = fmaxf(mx, e[j]); }
#pragma unroll
    for (int o = 16; o > 0; o >>= 1) mx = fmaxf(mx, __shfl_xor(mx, o, 32));
    float s = 0.0f;
#pragma unroll
    for (int j = 0; j < 8; ++j) { e[j] = __expf(e[j] - mx); s += e[j]; }
#pragma unroll
    for (int o = 16; o > 0; o >>= 1) s += __shfl_xor(s, o, 32);
    const float inv = 1.0f / s;
#pragma unroll
    for (int h = 0; h < 8; ++h) {
      const float vc = vs[dt][h][n] * inv;
#pragma unroll
      for (int j = 0; j < 8; ++j) oacc[h][j] += vc * e[j];
    }
  }

#pragma unroll
  for (int h = 0; h < 8; ++h)
#pragma unroll
    for (int j = 0; j < 8; ++j)
      oa[nh][dt][h][lane * 8 + j] = oacc[h][j];
  __syncthreads();

  for (int i = tid; i < 8192; i += 256) {
    int dt2 = i >> 11, h = (i >> 8) & 7, m = i & 255;
    A[base + (size_t)((h << 8) + d0 + dt2) * PDIM + m] = oa[0][dt2][h][m] + oa[1][dt2][h][m];
  }
}

// ---------------------------------------------------------------------------
extern "C" void kernel_launch(void* const* d_in, const int* in_sizes, int n_in,
                              void* d_out, int out_size, void* d_ws, size_t ws_size,
                              hipStream_t stream)
{
  (void)in_sizes; (void)n_in; (void)out_size; (void)ws_size;
  const float* x     = (const float*)d_in[0];
  const float* wq    = (const float*)d_in[1];
  const float* bq    = (const float*)d_in[2];
  const float* wk    = (const float*)d_in[3];
  const float* bk    = (const float*)d_in[4];
  const float* wv    = (const float*)d_in[5];
  const float* bv    = (const float*)d_in[6];
  const float* wf    = (const float*)d_in[7];
  const float* bf    = (const float*)d_in[8];
  const float* gamma = (const float*)d_in[9];
  float* out = (float*)d_out;

  const size_t TSZ = (size_t)BATCH * CDIM * PDIM;   // 4,194,304 floats (16.8 MB)
  float* Qb = (float*)d_ws;
  float* Kb = Qb + TSZ;
  float* Vb = Kb + TSZ;
  float* Ab = Vb + TSZ;

  dim3 gemmGrid(NTOT / 128, CDIM / 128);            // 16 x 16 workgroups
  proj_gemm_bf16x3<<<gemmGrid, 256, 0, stream>>>(wq, x, bq, nullptr, nullptr, Qb);
  proj_gemm_bf16x3<<<gemmGrid, 256, 0, stream>>>(wk, x, bk, nullptr, nullptr, Kb);
  proj_gemm_bf16x3<<<gemmGrid, 256, 0, stream>>>(wv, x, bv, nullptr, nullptr, Vb);
  attention_fused<<<dim3(PDIM / 4, BATCH), 256, 0, stream>>>(Qb, Kb, Vb, Ab);
  proj_gemm_bf16x3<<<gemmGrid, 256, 0, stream>>>(wf, Ab, bf, gamma, x, out);
}
